// LSTM_50508815401330
// MI455X (gfx1250) — matrix operand order repaired and checked by reading, not compiled, not run on hardware
//
#include <hip/hip_runtime.h>
#include <stdint.h>

// Problem constants (match reference)
#define B_TOT 4096
#define L_SEQ 256
#define H_DIM 32
#define D_DIM 2
#define GCOLS 128   // 4*H

typedef __attribute__((ext_vector_type(16))) _Float16 v16h;
typedef __attribute__((ext_vector_type(8)))  float    v8f;

union V16HQ { v16h v; uint4 q[2]; };

__device__ __forceinline__ float sigmoid_f(float x) {
    return __builtin_amdgcn_rcpf(1.0f + __expf(-x)); // v_exp_f32 + v_rcp_f32
}
#if __has_builtin(__builtin_amdgcn_tanhf)
__device__ __forceinline__ float tanh_f(float x) { return __builtin_amdgcn_tanhf(x); }
#else
__device__ __forceinline__ float tanh_f(float x) { return 2.0f * sigmoid_f(2.0f * x) - 1.0f; }
#endif
__device__ __forceinline__ float elu_f(float x) {
    return x > 0.0f ? x : (__expf(x) - 1.0f);
}

__global__ __launch_bounds__(32)
__attribute__((amdgpu_waves_per_eu(1)))
void lstm_wmma_kernel(const int*   __restrict__ x,    // [B, L] spins in {0,1}
                      const float* __restrict__ Wi,   // [D, 4H]
                      const float* __restrict__ Wh,   // [H, 4H]
                      const float* __restrict__ bh,   // [4H]
                      const float* __restrict__ Wo,   // [H, D]
                      const float* __restrict__ bo,   // [D]
                      float*       __restrict__ out)  // [B]
{
    // One wave32 per 16-row batch tile. All recurrent state stays resident.
    __shared__ unsigned char lds_spin[16 * L_SEQ];   // 4KB staged spins
    __shared__ _Float16      lds_whT[GCOLS * H_DIM]; // 8KB Wh^T f16: [col][k]
    __shared__ _Float16      lds_woT[16 * H_DIM];    // 1KB Wo^T f16 (cols>=2 zero)
    __shared__ _Float16      lds_h[16 * H_DIM];      // 1KB h_new relayout bounce

    const int lane  = threadIdx.x;     // 0..31
    const int n     = lane & 15;       // column within 16-wide tile
    const int halfS = lane >> 4;       // 0: rows 0-7 / K-lo ; 1: rows 8-15 / K-hi
    const int r0    = blockIdx.x * 16; // first global batch row of this tile

    // ---- Stage spins (int32 -> u8) into LDS: 16 rows x 256 steps ----------
    for (int i = lane; i < (16 * L_SEQ) / 4; i += 32) {
        int e  = i * 4;
        int ml = e >> 8;       // local row
        int t0 = e & 255;      // step
        const int4 v = *(const int4*)(x + (size_t)(r0 + ml) * L_SEQ + t0);
        uint32_t pk = (uint32_t)(v.x & 0xFF) | ((uint32_t)(v.y & 0xFF) << 8)
                    | ((uint32_t)(v.z & 0xFF) << 16) | ((uint32_t)(v.w & 0xFF) << 24);
        *(uint32_t*)(lds_spin + e) = pk;
    }
    // ---- Stage Wh^T and Wo^T into LDS as f16 (one-time, coalesced) --------
    for (int i = lane; i < GCOLS * H_DIM; i += 32) {
        int col = i & (GCOLS - 1);
        int k   = i >> 7;               // /128
        lds_whT[col * H_DIM + k] = (_Float16)Wh[k * GCOLS + col];
    }
    for (int i = lane; i < 16 * H_DIM; i += 32) {
        int col = i & 15;
        int k   = i >> 4;
        lds_woT[col * H_DIM + k] = (col < D_DIM) ? (_Float16)Wo[k * D_DIM + col]
                                                 : (_Float16)0.0f;
    }
    asm volatile("s_wait_dscnt 0" ::: "memory"); // single-wave: DS in-order, but pin it

    // ---- Load WMMA B-layout registers from LDS (ds_load_b128 pairs) -------
    // B 32x16 f16: lane n holds column N=n; lanes<16 K=0..15, lanes>=16 K=16..31.
    V16HQ bWh[8];
#pragma unroll
    for (int nt = 0; nt < 8; ++nt) {
        const _Float16* p = &lds_whT[(nt * 16 + n) * H_DIM + 16 * halfS];
        bWh[nt].q[0] = *(const uint4*)(p);
        bWh[nt].q[1] = *(const uint4*)(p + 8);
    }
    V16HQ bWo;
    {
        const _Float16* p = &lds_woT[n * H_DIM + 16 * halfS];
        bWo.q[0] = *(const uint4*)(p);
        bWo.q[1] = *(const uint4*)(p + 8);
    }

    // ---- Per-lane gate-column constants: base = bh+Wi[0], diff = Wi[1]-Wi[0]
    float b0r[8], dWr[8], bhr[8];
#pragma unroll
    for (int nt = 0; nt < 8; ++nt) {
        int col  = nt * 16 + n;
        float w0 = Wi[col];
        float w1 = Wi[GCOLS + col];
        float bb = bh[col];
        bhr[nt] = bb;
        b0r[nt] = bb + w0;
        dWr[nt] = w1 - w0;
    }
    const float boc = (n < D_DIM) ? bo[n] : 0.0f;

    // ---- Recurrent state ---------------------------------------------------
    float cst[16];   // c for (hsel, row j): index hsel*8 + j
    float lacc[8];   // log-prob accumulators (valid on lanes 0 and 16)
#pragma unroll
    for (int i = 0; i < 16; ++i) cst[i] = 0.0f;
#pragma unroll
    for (int j = 0; j < 8; ++j) lacc[j] = 0.0f;
    v16h aPrev = {}; // h0 = 0 in A-layout

    // ---- Time loop: 256 strictly sequential steps -------------------------
#pragma unroll 1
    for (int t = 0; t < L_SEQ; ++t) {
        // C init: gates = one_hot(x[t-1]) @ Wi + bh  (zeros input at t=0)
        v8f acc[8];
        if (t == 0) {
#pragma unroll
            for (int nt = 0; nt < 8; ++nt)
#pragma unroll
                for (int j = 0; j < 8; ++j) acc[nt][j] = bhr[nt];
        } else {
            float spf[8];
#pragma unroll
            for (int j = 0; j < 8; ++j)
                spf[j] = (float)lds_spin[(j + 8 * halfS) * L_SEQ + (t - 1)];
#pragma unroll
            for (int nt = 0; nt < 8; ++nt)
#pragma unroll
                for (int j = 0; j < 8; ++j)
                    acc[nt][j] = fmaf(spf[j], dWr[nt], b0r[nt]);
        }

        // gates += h_{t-1} @ Wh  : 8 WMMAs cover the 16x128 gate block, K=32
#pragma unroll
        for (int nt = 0; nt < 8; ++nt)
            acc[nt] = __builtin_amdgcn_wmma_f32_16x16x32_f16(
                false, aPrev, false, bWh[nt].v, (short)0, acc[nt], false, false);

        // Elementwise LSTM cell update; h_new (f16) -> LDS row-major 16x32
#pragma unroll
        for (int hsel = 0; hsel < 2; ++hsel) {
#pragma unroll
            for (int j = 0; j < 8; ++j) {
                float gi = acc[0 + hsel][j];   // i: cols 0..31
                float gf = acc[2 + hsel][j];   // f: cols 32..63
                float gg = acc[4 + hsel][j];   // g: cols 64..95
                float go = acc[6 + hsel][j];   // o: cols 96..127
                float c  = sigmoid_f(gf) * cst[hsel * 8 + j]
                         + sigmoid_f(gi) * tanh_f(gg);
                float h  = sigmoid_f(go) * tanh_f(c);
                cst[hsel * 8 + j] = c;
                lds_h[(j + 8 * halfS) * H_DIM + (n + 16 * hsel)] = (_Float16)h;
            }
        }
        asm volatile("s_wait_dscnt 0" ::: "memory");

        // Reload h_new in WMMA A-layout: lane m=n; K {0..15} / {16..31}
        V16HQ a;
        {
            const _Float16* hp = &lds_h[n * H_DIM + halfS * 16];
            a.q[0] = *(const uint4*)(hp);       // ds_load_b128
            a.q[1] = *(const uint4*)(hp + 8);   // ds_load_b128
        }

        // out = elu(h_new @ Wo + bo): one WMMA, columns 0..1 meaningful
        v8f oacc;
#pragma unroll
        for (int j = 0; j < 8; ++j) oacc[j] = boc;
        oacc = __builtin_amdgcn_wmma_f32_16x16x32_f16(
            false, a.v, false, bWo.v, (short)0, oacc, false, false);

        // Exchange col0<->col1 with full EXEC (keeps WMMA region uniform)
        float oth[8];
#pragma unroll
        for (int j = 0; j < 8; ++j) oth[j] = __shfl_xor(oacc[j], 1, 32);

        // Stable 2-way log-softmax gather, on lanes 0 and 16 only
        if (n == 0) {
#pragma unroll
            for (int j = 0; j < 8; ++j) {
                int   s   = lds_spin[(j + 8 * halfS) * L_SEQ + t];
                float o0  = elu_f(oacc[j]);
                float o1  = elu_f(oth[j]);
                float mx  = fmaxf(o0, o1);
                float lse = mx + __logf(__expf(o0 - mx) + __expf(o1 - mx));
                lacc[j]  += (s ? o1 : o0) - lse;
            }
        }
        aPrev = a.v;
    }

    // ---- Epilogue: LOG_PROB_FACTOR * sum_t logp ---------------------------
    if (n == 0) {
#pragma unroll
        for (int j = 0; j < 8; ++j)
            out[r0 + 8 * halfS + j] = 0.5f * lacc[j];
    }
}

extern "C" void kernel_launch(void* const* d_in, const int* in_sizes, int n_in,
                              void* d_out, int out_size, void* d_ws, size_t ws_size,
                              hipStream_t stream) {
    (void)in_sizes; (void)n_in; (void)out_size; (void)d_ws; (void)ws_size;
    const int*   x  = (const int*)d_in[0];
    const float* Wi = (const float*)d_in[1];
    const float* Wh = (const float*)d_in[2];
    const float* bh = (const float*)d_in[3];
    const float* Wo = (const float*)d_in[4];
    const float* bo = (const float*)d_in[5];
    float* out = (float*)d_out;

    dim3 grid(B_TOT / 16);  // 256 waves, one 16-row batch tile each
    dim3 block(32);         // one wave32
    lstm_wmma_kernel<<<grid, block, 0, stream>>>(x, Wi, Wh, bh, Wo, bo, out);
}
